// self_attention_89369679495789
// MI455X (gfx1250) — compile-verified, matrix operand order checked by reading, not run on hardware
//
#include <hip/hip_runtime.h>
#include <stdint.h>

// MI455X / gfx1250 fused "self-attention-like" kernel.
// Pipeline: prep (bf16 convert + colsum + Wt + opinion weights) ->
//           gemm_xw (x@W+b via v_wmma_f32_16x16x32_bf16, TDM-staged A tile) ->
//           attn_fused (flash-style: scores, exp(tanh) epilogue, out accumulation,
//                       TDM-staged key chunks, LDS reduction, WMMA both GEMMs).
// Workspace requirement: ~129 MB (xb 64MB + xt 64MB + Wt 0.5MB + ow 0.25MB + colsum 0.125MB).

#define B_ 64
#define S_ 1024
#define D_ 512
#define EPSF 1e-10f

typedef __attribute__((ext_vector_type(16))) __bf16 v16bf;
typedef __attribute__((ext_vector_type(8)))  float  v8f;
typedef __attribute__((ext_vector_type(4)))  unsigned int u32x4;
typedef __attribute__((ext_vector_type(8)))  int          i32x8;
typedef __attribute__((ext_vector_type(4)))  int          i32x4;

// ---------------------------------------------------------------------------
// Tensor Data Mover: async DMA of `rows` x 512 bf16 row-major tile -> LDS.
// Builds a D# per CDNA5 ISA (group0: count/lds/global/type,
// group1: data_size=2B, tensor_dim0=512, tensor_dim1=rows, tile=512 x rows,
// dim0_stride=512). Groups 2/3 zero (2D tensor). Tracked on TENSORcnt.
// Issue from a single wave; TDM ignores EXEC.
// This toolchain's builtin is the 6-arg form:
//   (u32x4 g0, i32x8 g1, i32x4 g2, i32x4 g3, i32x8 g4, i32 cpol)
// ---------------------------------------------------------------------------
__device__ __forceinline__ void tdm_load_rows_bf16(void* lds_dst, const void* gsrc,
                                                   int rows) {
  unsigned lds_off = (unsigned)(uintptr_t)lds_dst;
  unsigned long long ga = (unsigned long long)(uintptr_t)gsrc;
  u32x4 g0;
  g0[0] = 1u;                                            // count=1 (valid user D#)
  g0[1] = lds_off;                                       // lds_addr
  g0[2] = (unsigned)(ga & 0xFFFFFFFFull);                // global_addr[31:0]
  g0[3] = (unsigned)((ga >> 32) & 0x01FFFFFFull)         // global_addr[56:32]
          | 0x80000000u;                                 // type=2 ("image")
  i32x8 g1;
  g1[0] = 0x00010000;               // data_size=1 -> 2-byte elements; mask=0
  g1[1] = (int)(D_ << 16);          // tensor_dim0[15:0] @ bits 63:48  (=512)
  g1[2] = (int)(rows << 16);        // tensor_dim1[15:0] @ bits 95:80 (dim0 hi=0)
  g1[3] = (int)(D_ << 16);          // tile_dim0 @ bits 127:112 (=512)
  g1[4] = rows;                     // tile_dim1 (tile_dim2 = 0)
  g1[5] = D_;                       // tensor_dim0_stride[31:0] (=512 elements)
  g1[6] = 0;                        // stride hi / dim1_stride lo
  g1[7] = 0;
  i32x4 z4; z4[0] = 0; z4[1] = 0; z4[2] = 0; z4[3] = 0;
  i32x8 z8; z8[0] = 0; z8[1] = 0; z8[2] = 0; z8[3] = 0;
  z8[4] = 0; z8[5] = 0; z8[6] = 0; z8[7] = 0;
  __builtin_amdgcn_tensor_load_to_lds(g0, g1, z4, z4, z8, 0);
}

// ---------------------------------------------------------------------------
// prep: fp32 x -> bf16 xb, and per-batch column sums  colsum[b,d] = sum_t x[b,t,d]
// grid (64, 4) x 256 threads; 2 threads (t-halves) per column.
// ---------------------------------------------------------------------------
__global__ void __launch_bounds__(256)
prep_xb_colsum(const float* __restrict__ x, __bf16* __restrict__ xb,
               float* __restrict__ colsum) {
  int b = blockIdx.x;
  int tid = threadIdx.x;
  int col = blockIdx.y * 128 + (tid & 127);
  int seg = tid >> 7;
  const float* xp = x + (size_t)b * S_ * D_;
  __bf16* xbp = xb + (size_t)b * S_ * D_;
  float sum = 0.f;
  for (int t = seg * 512; t < seg * 512 + 512; ++t) {
    float v = xp[(size_t)t * D_ + col];
    sum += v;
    xbp[(size_t)t * D_ + col] = (__bf16)v;
  }
  __shared__ float red[256];
  red[tid] = sum;
  __syncthreads();
  if (tid < 128) colsum[b * D_ + col] = red[tid] + red[tid + 128];
}

// W (fp32, row-major [k][c]) -> Wt (bf16, [c][k]) so WMMA B loads are contiguous.
__global__ void __launch_bounds__(256)
prep_wt(const float* __restrict__ W, __bf16* __restrict__ Wt) {
  int id0 = (blockIdx.x * 256 + threadIdx.x) * 4;
#pragma unroll
  for (int i = 0; i < 4; ++i) {
    int id = id0 + i;
    int c = id >> 9, k = id & 511;
    Wt[(size_t)c * D_ + k] = (__bf16)W[(size_t)k * D_ + c];
  }
}

// opinion weights  ow[b,t] = gp*(g1+g2) + (1-gp)*(p3+p4)
__global__ void __launch_bounds__(256)
prep_ow(const float* __restrict__ go, const float* __restrict__ po,
        const float* __restrict__ gp, float* __restrict__ ow) {
  int b = blockIdx.x;
  for (int t = threadIdx.x; t < S_; t += 256) {
    float g = go[((size_t)b * S_ + t) * 3 + 1] + go[((size_t)b * S_ + t) * 3 + 2];
    float p = po[((size_t)b * S_ + t) * 5 + 3] + po[((size_t)b * S_ + t) * 5 + 4];
    float q = gp[b * S_ + t];
    ow[b * S_ + t] = q * g + (1.f - q) * p;
  }
}

// ---------------------------------------------------------------------------
// gemm_xw: x_tran = x @ W + b  (bf16 in, fp32 accumulate, bf16 out)
// One 16-row tile per block (4096 blocks); 8 waves each own 64 output columns.
// A tile TDM-staged to LDS; B (Wt) contiguous 32B/lane from L2-resident buffer.
// ---------------------------------------------------------------------------
__global__ void __launch_bounds__(256)
gemm_xw(const __bf16* __restrict__ xb, const __bf16* __restrict__ Wt,
        const float* __restrict__ bias, __bf16* __restrict__ xt) {
  __shared__ __align__(32) __bf16 arow[16][D_];  // 16 KB

  int bid = blockIdx.x;
  int b = bid >> 6, rt = bid & 63;
  int row0 = rt * 16;
  int wave = threadIdx.x >> 5, lane = threadIdx.x & 31;

  if (wave == 0) {
    tdm_load_rows_bf16(&arow[0][0], xb + ((size_t)b * S_ + row0) * D_, 16);
    __builtin_amdgcn_s_wait_tensorcnt(0);
  }
  __syncthreads();

  int c0 = wave * 64;
  int mrow = lane & 15;
  int kh = (lane >> 4) << 4;    // per-lane K half-slice (layout approximation)
  int mBase = (lane >> 4) << 3; // C/D layout: lanes 16-31 hold M+8

  v8f acc[4] = {};
#pragma unroll 4
  for (int k0 = 0; k0 < D_; k0 += 32) {
    v16bf av = *(const v16bf*)&arow[mrow][k0 + kh];
#pragma unroll
    for (int j = 0; j < 4; ++j) {
      const __bf16* bp = Wt + (size_t)(c0 + j * 16 + mrow) * D_ + k0 + kh;
      v16bf bv = *(const v16bf*)bp;
      acc[j] = __builtin_amdgcn_wmma_f32_16x16x32_bf16(
          false, av, false, bv, (short)0, acc[j], false, false);
    }
  }

#pragma unroll
  for (int j = 0; j < 4; ++j) {
    int col = c0 + j * 16 + (lane & 15);
    float bb = bias[col];
#pragma unroll
    for (int r = 0; r < 8; ++r) {
      int m = r + mBase;
      xt[((size_t)b * S_ + row0 + m) * D_ + col] = (__bf16)(acc[j][r] + bb);
    }
  }
}

// ---------------------------------------------------------------------------
// attn_fused: for one (batch, 16-query-row) tile, stream 32-key chunks:
//   scores(16x32) = x_tran_tile @ x_chunk^T   (8 waves split K=512, LDS reduce)
//   epilogue: loc decay * opinion * exp(tanh), zero diag, rowsum accumulation
//   out_acc(16x512) += weights(16x32) @ x_chunk(32x512)   (waves split D)
// Finalize: out = out_acc / rowsum + EPS * colsum.
// LDS ~51 KB/block.
// ---------------------------------------------------------------------------
__global__ void __launch_bounds__(256)
attn_fused(const __bf16* __restrict__ xt, const __bf16* __restrict__ xb,
           const float* __restrict__ ow, const float* __restrict__ colsum,
           float* __restrict__ out) {
  __shared__ __align__(32) __bf16 a_xt[16][D_];    // 16 KB  query-tile x_tran
  __shared__ __align__(32) __bf16 xch[32][D_];     // 32 KB  key chunk
  __shared__ __align__(32) float  score32[16][32]; //  2 KB
  __shared__ __align__(32) __bf16 score_a[16][32]; //  1 KB  bf16 A for 2nd GEMM
  __shared__ float rowsum[16];

  int bid = blockIdx.x;
  int b = bid >> 6, rt = bid & 63;
  int i0 = rt * 16;
  int wave = threadIdx.x >> 5, lane = threadIdx.x & 31;
  int tid = threadIdx.x;

  if (wave == 0)
    tdm_load_rows_bf16(&a_xt[0][0], xt + ((size_t)b * S_ + i0) * D_, 16);
  if (tid < 16) rowsum[tid] = 0.f;
  v8f accO[4] = {};  // this wave's 64 D-columns of the output tile
  if (wave == 0) __builtin_amdgcn_s_wait_tensorcnt(0);
  __syncthreads();

  int mrow = lane & 15;
  int kh = (lane >> 4) << 4;
  int mBase = (lane >> 4) << 3;

  for (int ct = 0; ct < 32; ++ct) {
    int t0 = ct * 32;
    if (wave == 0) {
      tdm_load_rows_bf16(&xch[0][0], xb + ((size_t)b * S_ + t0) * D_, 32);
      __builtin_amdgcn_s_wait_tensorcnt(0);
    }
    ((float*)score32)[tid] = 0.f;
    ((float*)score32)[tid + 256] = 0.f;
    __syncthreads();

    // Phase A: partial scores over this wave's K slice of 64 (2 x K=32 WMMA)
#pragma unroll
    for (int nt = 0; nt < 2; ++nt) {
      v8f acc = {};
#pragma unroll
      for (int kk = 0; kk < 2; ++kk) {
        int k0 = wave * 64 + kk * 32;
        v16bf av = *(const v16bf*)&a_xt[mrow][k0 + kh];
        v16bf bv = *(const v16bf*)&xch[nt * 16 + mrow][k0 + kh];
        acc = __builtin_amdgcn_wmma_f32_16x16x32_bf16(
            false, av, false, bv, (short)0, acc, false, false);
      }
#pragma unroll
      for (int r = 0; r < 8; ++r)
        atomicAdd(&score32[r + mBase][nt * 16 + (lane & 15)], acc[r]);
    }
    __syncthreads();

    // Phase B: elementwise epilogue; wave w owns score rows m=w and m=w+8
#pragma unroll
    for (int half = 0; half < 2; ++half) {
      int m = wave + half * 8;
      int i = i0 + m;
      int t = t0 + lane;
      float s = score32[m][lane];
      float diff = fabsf((float)(i - t));
      float loc = (i == t) ? 0.f : 1.f / (diff + EPSF);
      float wgt = s * loc * ow[b * S_ + t];
      float e = (i == t) ? 0.f : __expf(tanhf(wgt));
      score_a[m][lane] = (__bf16)e;
      float r = e;  // wave32 reduction for rowsum
      for (int off = 16; off > 0; off >>= 1) r += __shfl_down(r, off);
      if (lane == 0) rowsum[m] += r;
    }
    __syncthreads();

    // Phase C: out_acc += weights(16x32) @ x_chunk; B gathered column-wise
    {
      v16bf av = *(const v16bf*)&score_a[mrow][kh];
#pragma unroll
      for (int j = 0; j < 4; ++j) {
        int dcol = wave * 64 + j * 16 + (lane & 15);
        v16bf bv;
#pragma unroll
        for (int kk = 0; kk < 16; ++kk) bv[kk] = xch[kh + kk][dcol];
        accO[j] = __builtin_amdgcn_wmma_f32_16x16x32_bf16(
            false, av, false, bv, (short)0, accO[j], false, false);
      }
    }
    __syncthreads();  // xch / score tiles reused next chunk
  }

  // Finalize: normalize + EPS * colsum  (out = acc/rowsum + EPS * sum_t x[t,d])
#pragma unroll
  for (int j = 0; j < 4; ++j) {
    int dcol = wave * 64 + j * 16 + (lane & 15);
    float cs = colsum[b * D_ + dcol];
#pragma unroll
    for (int r = 0; r < 8; ++r) {
      int m = r + mBase;
      float val = accO[j][r] / rowsum[m] + EPSF * cs;
      out[((size_t)b * S_ + i0 + m) * D_ + dcol] = val;
    }
  }
}

// ---------------------------------------------------------------------------
extern "C" void kernel_launch(void* const* d_in, const int* in_sizes, int n_in,
                              void* d_out, int out_size, void* d_ws, size_t ws_size,
                              hipStream_t stream) {
  const float* x    = (const float*)d_in[0];
  const float* go   = (const float*)d_in[1];
  const float* po   = (const float*)d_in[2];
  const float* gp   = (const float*)d_in[3];
  const float* W    = (const float*)d_in[4];
  const float* bias = (const float*)d_in[5];
  float* out = (float*)d_out;

  char* ws = (char*)d_ws;
  __bf16* xb = (__bf16*)ws;  ws += (size_t)B_ * S_ * D_ * 2;  // 64 MB
  __bf16* xt = (__bf16*)ws;  ws += (size_t)B_ * S_ * D_ * 2;  // 64 MB
  __bf16* Wt = (__bf16*)ws;  ws += (size_t)D_ * D_ * 2;       // 512 KB
  float* owv = (float*)ws;   ws += (size_t)B_ * S_ * 4;       // 256 KB
  float* cs  = (float*)ws;   ws += (size_t)B_ * D_ * 4;       // 128 KB

  prep_xb_colsum<<<dim3(B_, 4), 256, 0, stream>>>(x, xb, cs);
  prep_wt<<<dim3(256), 256, 0, stream>>>(W, Wt);
  prep_ow<<<dim3(B_), 256, 0, stream>>>(go, po, gp, owv);
  gemm_xw<<<dim3(B_ * S_ / 16), 256, 0, stream>>>(xb, Wt, bias, xt);
  attn_fused<<<dim3(B_ * S_ / 16), 256, 0, stream>>>(xt, xb, owv, cs, out);
}
